// DSAFTMAELoss_7799660609715
// MI455X (gfx1250) — compile-verified, matrix operand order checked by reading, not run on hardware
//
#include <hip/hip_runtime.h>
#include <math.h>

#define NN   8192
#define EPSF 1e-32f

typedef __attribute__((ext_vector_type(16))) _Float16 v16h;
typedef __attribute__((ext_vector_type(8)))  float    v8f;

// ---------------------------------------------------------------------------
// Kernel 1: e[i] = log(dur[i]+eps) - theta[i]; LDS bitonic sort (key + index);
// outputs e_sorted[] and the inverse permutation inv[] (inv[idx[j]] = j).
// ---------------------------------------------------------------------------
__global__ void __launch_bounds__(1024)
k_sort(const float* __restrict__ theta, const float* __restrict__ dur,
       float* __restrict__ e_sorted, int* __restrict__ inv) {
  __shared__ float key[NN];   // 32 KB
  __shared__ int   val[NN];   // 32 KB  (64 KB total, fits WGP LDS)
  const int t = threadIdx.x;
  for (int i = t; i < NN; i += 1024) {
    key[i] = logf(dur[i] + EPSF) - theta[i];
    val[i] = i;
  }
  __syncthreads();
  for (unsigned k = 2; k <= NN; k <<= 1) {
    for (unsigned j = k >> 1; j > 0; j >>= 1) {
      for (unsigned i = (unsigned)t; i < NN; i += 1024) {
        unsigned p = i ^ j;
        if (p > i) {
          bool up = ((i & k) == 0);
          float a = key[i], b = key[p];
          if ((a > b) == up) {
            key[i] = b; key[p] = a;
            int tv = val[i]; val[i] = val[p]; val[p] = tv;
          }
        }
      }
      __syncthreads();
    }
  }
  for (int i = t; i < NN; i += 1024) {
    e_sorted[i] = key[i];
    inv[val[i]] = i;          // inverse permutation
  }
}

// ---------------------------------------------------------------------------
// Kernel 2: replicate the reference's inverse-permutation gather:
//   x_s[i] = x[inv[i]]
// ---------------------------------------------------------------------------
__global__ void __launch_bounds__(256)
k_gather(const float* __restrict__ theta, const float* __restrict__ dur,
         const int* __restrict__ ev, const int* __restrict__ inv,
         float* __restrict__ events_s, float* __restrict__ logd_s,
         float* __restrict__ theta_s) {
  int i = blockIdx.x * 256 + threadIdx.x;
  int s = inv[i];
  events_s[i] = (float)ev[s];
  logd_s[i]   = logf(dur[s] + EPSF);
  theta_s[i]  = theta[s];
}

// ---------------------------------------------------------------------------
// Kernel 3: n_at_risk[i] = #{ j : e_s[j] >= e_s[i] } via WMMA masked count.
// A tile = 0/1 mask (exact in f16); B has ones only in column N=0, so
// D[:,0] accumulates exact integer row counts in f32.
// The column stream e_sorted[] is identical for every lane/wave, so it is
// staged once per block into LDS; inner-loop reads become broadcast
// ds_load's instead of redundant global_load_b128 + loadcnt waits.
// One wave per 16-row strip; K loop over all 8192 columns (256 WMMAs/wave).
// ---------------------------------------------------------------------------
__global__ void __launch_bounds__(256)
k_at_risk_wmma(const float* __restrict__ e_sorted, float* __restrict__ nar) {
  __shared__ float cols[NN];   // 32 KB column stage, shared by all 8 waves
  for (int i = threadIdx.x; i < NN; i += 256) cols[i] = e_sorted[i];
  __syncthreads();

  const int lane = threadIdx.x & 31;
  const int wave = threadIdx.x >> 5;
  const int i0   = (blockIdx.x * 8 + wave) * 16;
  const int m    = lane & 15;
  const float r  = cols[i0 + m];
  const _Float16 h1 = (_Float16)1.0f;
  const _Float16 h0 = (_Float16)0.0f;

  // B (32x16): column 0 all ones, others zero. Lane supplies column N=lane&15.
  v16h bcol;
  {
    _Float16 bv = (m == 0) ? h1 : h0;
#pragma unroll
    for (int q = 0; q < 16; ++q) bcol[q] = bv;
  }

  v8f acc = {};
  // A layout: lanes 0-15 -> a[0..7]=K(0..7), a[8..15]=K(16..23);
  //           lanes 16-31 -> K(8..15) / K(24..31)
  const int cbase = (lane < 16) ? 0 : 8;
  for (int j0 = 0; j0 < NN; j0 += 32) {
    v16h a;
#pragma unroll
    for (int q = 0; q < 8; ++q) {
      float c0 = cols[j0 + cbase + q];
      float c1 = cols[j0 + cbase + 16 + q];
      a[q]     = (r <= c0) ? h1 : h0;
      a[q + 8] = (r <= c1) ? h1 : h0;
    }
    acc = __builtin_amdgcn_wmma_f32_16x16x32_f16(false, a, false, bcol,
                                                 (short)0, acc, false, false);
  }
  // D column 0: lane 0 holds rows 0..7 in acc[0..7]; lane 16 holds rows 8..15.
  if (m == 0) {
    const int ro = (lane == 0) ? 0 : 8;
#pragma unroll
    for (int q = 0; q < 8; ++q) nar[i0 + ro + q] = acc[q];
  }
}

// ---------------------------------------------------------------------------
// Kernel 4: scans + final loss.
//   v[j]   = |1 - events_s[j]/n_at_risk[j]|
//   P      = exclusive prefix product of v        -> surv[i] = P[lo[i]]
//   w[i]   = e_s[i] * d_cdf[i],  d_cdf from F=|surv-1|
//   S      = suffix sum of w                      -> cond_E[i] = S[lo[i]]/surv[i]
//   loss   = mean |events_s*(logd_s - cond_E - theta_s) + cond_E|
// One 1024-thread block, 8 elements/thread, Hillis-Steele block scans.
// ---------------------------------------------------------------------------
__global__ void __launch_bounds__(1024)
k_scan_finish(const float* __restrict__ e_sorted,
              const float* __restrict__ events_s,
              const float* __restrict__ logd_s,
              const float* __restrict__ theta_s,
              const float* __restrict__ nar,
              float* __restrict__ surv_g,
              float* __restrict__ out) {
  __shared__ float buf[NN];      // 32 KB (holds P, then S)
  __shared__ float part[1024];
  const int t  = threadIdx.x;
  const int q0 = t * 8;

  // ---- exclusive prefix product of v -> buf
  float lp[8];
  {
    float run = 1.0f;
#pragma unroll
    for (int q = 0; q < 8; ++q) {
      int i = q0 + q;
      float v = fabsf(1.0f - events_s[i] / nar[i]);
      run *= v;
      lp[q] = run;               // inclusive within chunk
    }
  }
  part[t] = lp[7];
  __syncthreads();
  for (int off = 1; off < 1024; off <<= 1) {
    float x = part[t];
    float y = (t >= off) ? part[t - off] : 1.0f;
    __syncthreads();
    part[t] = x * y;
    __syncthreads();
  }
  float offp = (t == 0) ? 1.0f : part[t - 1];
  buf[q0] = offp;
#pragma unroll
  for (int q = 1; q < 8; ++q) buf[q0 + q] = offp * lp[q - 1];
  __syncthreads();

  // ---- surv[i] = P[lo[i]]
  int   lo[8];
  float sv[8];
#pragma unroll
  for (int q = 0; q < 8; ++q) {
    int i = q0 + q;
    lo[q] = NN - (int)(nar[i] + 0.5f);   // counts are exact integers
    sv[q] = buf[lo[q]];
    surv_g[i] = sv[q];
  }
  __syncthreads();

  // ---- w[i] = e_s[i] * d_cdf[i]; suffix-sum S -> buf (overwrites P)
  float ls[8];
  {
    float w[8];
#pragma unroll
    for (int q = 0; q < 8; ++q) {
      int i = q0 + q;
      float Fi  = fabsf(sv[q] - 1.0f);
      float nxt = (i < NN - 1) ? fabsf(surv_g[i + 1] - 1.0f) : 1.0f;
      w[q] = e_sorted[i] * (nxt - Fi);
    }
    float run = 0.0f;
#pragma unroll
    for (int q = 7; q >= 0; --q) { run += w[q]; ls[q] = run; }
  }
  part[t] = ls[0];
  __syncthreads();
  for (int off = 1; off < 1024; off <<= 1) {
    float x = part[t];
    float y = (t + off < 1024) ? part[t + off] : 0.0f;
    __syncthreads();
    part[t] = x + y;
    __syncthreads();
  }
  float soff = (t < 1023) ? part[t + 1] : 0.0f;
#pragma unroll
  for (int q = 0; q < 8; ++q) buf[q0 + q] = ls[q] + soff;
  __syncthreads();

  // ---- loss terms + reduction
  float acc = 0.0f;
#pragma unroll
  for (int q = 0; q < 8; ++q) {
    int i = q0 + q;
    float condE = buf[lo[q]] / sv[q];
    float term  = events_s[i] * (logd_s[i] - condE - theta_s[i]) + condE;
    acc += fabsf(term);
  }
  part[t] = acc;
  __syncthreads();
  for (int s = 512; s > 0; s >>= 1) {
    if (t < s) part[t] += part[t + s];
    __syncthreads();
  }
  if (t == 0) out[0] = part[0] * (1.0f / NN);
}

// ---------------------------------------------------------------------------
extern "C" void kernel_launch(void* const* d_in, const int* in_sizes, int n_in,
                              void* d_out, int out_size, void* d_ws, size_t ws_size,
                              hipStream_t stream) {
  (void)in_sizes; (void)n_in; (void)out_size; (void)ws_size;
  const float* theta = (const float*)d_in[0];
  const float* dur   = (const float*)d_in[1];
  const int*   ev    = (const int*)d_in[2];

  float* ws       = (float*)d_ws;
  float* e_sorted = ws + 0 * NN;
  int*   inv      = (int*)(ws + 1 * NN);
  float* events_s = ws + 2 * NN;
  float* logd_s   = ws + 3 * NN;
  float* theta_s  = ws + 4 * NN;
  float* nar      = ws + 5 * NN;
  float* surv     = ws + 6 * NN;

  hipLaunchKernelGGL(k_sort, dim3(1), dim3(1024), 0, stream,
                     theta, dur, e_sorted, inv);
  hipLaunchKernelGGL(k_gather, dim3(NN / 256), dim3(256), 0, stream,
                     theta, dur, ev, inv, events_s, logd_s, theta_s);
  hipLaunchKernelGGL(k_at_risk_wmma, dim3(NN / 16 / 8), dim3(256), 0, stream,
                     e_sorted, nar);
  hipLaunchKernelGGL(k_scan_finish, dim3(1), dim3(1024), 0, stream,
                     e_sorted, events_s, logd_s, theta_s, nar, surv,
                     (float*)d_out);
}
